// SequenceLabelingLSTM_CRF_84189948936844
// MI455X (gfx1250) — compile-verified, matrix operand order checked by reading, not run on hardware
//
#include <hip/hip_runtime.h>
#include <hip/hip_bf16.h>

// ---------------------------------------------------------------------------
// 2-layer LSTM + FC head for gfx1250 (MI455X), bf16 WMMA with fp32 accumulate.
//   B=64, S=2048, D=64, H=256, 4H=1024, C=8
// Pipeline (all on `stream`):
//   1) convert weights to bf16, precompute bias sums, zero barrier counters
//   2) pre0 = x @ Wih0^T + b0    (parallel WMMA GEMM, written [S,B,4H])
//   3) layer-0 recurrence         (persistent 16-WG WMMA kernel + prefetch)
//   4) pre1 = h1 @ Wih1^T + b1   (parallel WMMA GEMM, written [S,B,4H])
//   5) layer-1 recurrence
//   6) emissions = mask(h2) @ Wfc^T + bfc
// pre is stored time-major [S,B,4H] so each recurrence step reads one
// contiguous 256KB block (and prefetches the next step's block).
// ---------------------------------------------------------------------------

typedef __bf16 bf16;
typedef __attribute__((ext_vector_type(16))) __bf16 v16bf;
typedef __attribute__((ext_vector_type(8)))  float  v8f;

#define B_   64
#define S_   2048
#define D_   64
#define H_   256
#define G4H  1024
#define C_   8
#define NWG  16

// ---------------- workspace layout (bytes) ----------------
static constexpr size_t OFF_P    = 0;                                   // pre  [S,B,4H] f32 (reused both layers)
static constexpr size_t SZ_P     = (size_t)B_ * S_ * G4H * 4;           // 536,870,912
static constexpr size_t OFF_H1   = OFF_P + SZ_P;                        // h1   [B,S,H] bf16
static constexpr size_t SZ_HSEQ  = (size_t)B_ * S_ * H_ * 2;            // 67,108,864
static constexpr size_t OFF_H2   = OFF_H1 + SZ_HSEQ;                    // h2   [B,S,H] bf16
static constexpr size_t OFF_WHH0 = OFF_H2 + SZ_HSEQ;                    // [4H,H] bf16
static constexpr size_t OFF_WHH1 = OFF_WHH0 + (size_t)G4H * H_ * 2;
static constexpr size_t OFF_WIH0 = OFF_WHH1 + (size_t)G4H * H_ * 2;     // [4H,D] bf16
static constexpr size_t OFF_WIH1 = OFF_WIH0 + (size_t)G4H * D_ * 2;     // [4H,H] bf16
static constexpr size_t OFF_B0   = OFF_WIH1 + (size_t)G4H * H_ * 2;     // bih0+bhh0 f32
static constexpr size_t OFF_B1   = OFF_B0 + G4H * 4;
static constexpr size_t OFF_HST  = OFF_B1 + G4H * 4;                    // h state, double buffered, bf16
static constexpr size_t OFF_CNT  = OFF_HST + (size_t)2 * B_ * H_ * 2;   // barrier counters
static constexpr size_t WS_NEED  = OFF_CNT + 256;

// ---------------- helpers ----------------
__device__ __forceinline__ v8f wmma_bf16(v16bf a, v16bf b, v8f c) {
  return __builtin_amdgcn_wmma_f32_16x16x32_bf16(false, a, false, b, (short)0, c,
                                                 false, false);
}

// 16-bit A/B fragment (16x32 / 32x16), per ISA layout:
//   lane L (0..15): "row" = L, K = {koff..koff+7, 16+koff..16+koff+7}, koff = 8*(L>=16)
// For A: row = M.  For B (= W^T, W row-major [N][K]): row = N.
__device__ __forceinline__ v16bf load_frag_bf16(const bf16* tile, int stride, int lane) {
  const int r = lane & 15;
  const int koff = (lane >> 4) << 3;
  const bf16* p = tile + r * stride + koff;
  union { uint4 u[2]; v16bf v; } x;
  x.u[0] = *(const uint4*)(p);
  x.u[1] = *(const uint4*)(p + 16);
  return x.v;
}

__device__ __forceinline__ v16bf load_frag_f32(const float* tile, int stride, int lane) {
  const int r = lane & 15;
  const int koff = (lane >> 4) << 3;
  const float* p = tile + r * stride + koff;
  v16bf a;
  #pragma unroll
  for (int j = 0; j < 8; ++j) { a[j] = (bf16)p[j]; a[8 + j] = (bf16)p[16 + j]; }
  return a;
}

__device__ __forceinline__ float fast_sigmoid(float x) {
  x = fminf(fmaxf(x, -20.0f), 20.0f);
  return 1.0f / (1.0f + __expf(-x));
}
__device__ __forceinline__ float fast_tanh(float x) {
  x = fminf(fmaxf(x, -15.0f), 15.0f);
  float e = __expf(-2.0f * x);
  return (1.0f - e) / (1.0f + e);
}

// ---------------- tiny prep kernels ----------------
__global__ void cvt_bf16_kernel(const float* __restrict__ s, bf16* __restrict__ d, int n) {
  int i = blockIdx.x * blockDim.x + threadIdx.x;
  if (i < n) d[i] = (bf16)s[i];
}
__global__ void add_bias_kernel(const float* __restrict__ a, const float* __restrict__ b,
                                float* __restrict__ o, int n) {
  int i = blockIdx.x * blockDim.x + threadIdx.x;
  if (i < n) o[i] = a[i] + b[i];
}
__global__ void zero_u32_kernel(unsigned* __restrict__ p, int n) {
  int i = blockIdx.x * blockDim.x + threadIdx.x;
  if (i < n) p[i] = 0u;
}

// ---------------- input-projection GEMMs (parallel, WMMA) ----------------
// A row m = b*S + t (x / h1 are [B,S,*]).  Output written time-major:
//   pre[((t)*B + b)*4H + col].  Each wave: 1 M-tile x 4 N-tiles (A reused 4x).
__global__ __launch_bounds__(256) void pre_gemm_x_kernel(
    const float* __restrict__ x, const bf16* __restrict__ wih,
    const float* __restrict__ bias, float* __restrict__ pre) {
  const int tid = threadIdx.x, wid = tid >> 5, lane = tid & 31;
  const int mt = blockIdx.x >> 1;                 // 0..8191
  const int ntb = (blockIdx.x & 1) * 32 + wid * 4; // base N tile (4 per wave)
  const int n = lane & 15, mo = (lane >> 4) << 3;
  const int b = (mt * 16) >> 11;                  // batch of this M tile
  const int t0 = (mt * 16) & (S_ - 1);            // first time index

  v8f acc[4];
  #pragma unroll
  for (int q = 0; q < 4; ++q) {
    float bv = bias[(ntb + q) * 16 + n];
    #pragma unroll
    for (int r = 0; r < 8; ++r) acc[q][r] = bv;
  }
  #pragma unroll
  for (int kk = 0; kk < D_ / 32; ++kk) {
    v16bf a = load_frag_f32(x + (size_t)(mt * 16) * D_ + kk * 32, D_, lane);
    #pragma unroll
    for (int q = 0; q < 4; ++q) {
      v16bf bq = load_frag_bf16(wih + (size_t)((ntb + q) * 16) * D_ + kk * 32, D_, lane);
      acc[q] = wmma_bf16(a, bq, acc[q]);
    }
  }
  #pragma unroll
  for (int q = 0; q < 4; ++q) {
    #pragma unroll
    for (int r = 0; r < 8; ++r) {
      size_t row = (size_t)(t0 + mo + r) * B_ + b;   // time-major
      pre[row * G4H + (ntb + q) * 16 + n] = acc[q][r];
    }
  }
}

__global__ __launch_bounds__(256) void pre_gemm_h_kernel(
    const bf16* __restrict__ hseq, const bf16* __restrict__ wih,
    const float* __restrict__ bias, float* __restrict__ pre) {
  const int tid = threadIdx.x, wid = tid >> 5, lane = tid & 31;
  const int mt = blockIdx.x >> 1;
  const int ntb = (blockIdx.x & 1) * 32 + wid * 4;
  const int n = lane & 15, mo = (lane >> 4) << 3;
  const int b = (mt * 16) >> 11;
  const int t0 = (mt * 16) & (S_ - 1);

  v8f acc[4];
  #pragma unroll
  for (int q = 0; q < 4; ++q) {
    float bv = bias[(ntb + q) * 16 + n];
    #pragma unroll
    for (int r = 0; r < 8; ++r) acc[q][r] = bv;
  }
  #pragma unroll
  for (int kk = 0; kk < H_ / 32; ++kk) {
    v16bf a = load_frag_bf16(hseq + (size_t)(mt * 16) * H_ + kk * 32, H_, lane);
    #pragma unroll
    for (int q = 0; q < 4; ++q) {
      v16bf bq = load_frag_bf16(wih + (size_t)((ntb + q) * 16) * H_ + kk * 32, H_, lane);
      acc[q] = wmma_bf16(a, bq, acc[q]);
    }
  }
  #pragma unroll
  for (int q = 0; q < 4; ++q) {
    #pragma unroll
    for (int r = 0; r < 8; ++r) {
      size_t row = (size_t)(t0 + mo + r) * B_ + b;
      pre[row * G4H + (ntb + q) * 16 + n] = acc[q][r];
    }
  }
}

// ---------------- persistent recurrence kernel ----------------
// 16 WGs; WG w owns hidden units [w*16, w*16+16) => N-tiles {g*H + w*16} for
// gates g=0..3.  Whh slice (64 rows x 256 K, 32KB bf16) lives in LDS all 2048
// steps.  h state double-buffered bf16 in global.  pre is time-major, so each
// step reads a contiguous 256KB block; next step's block is prefetched.
// Grid barrier: monotonic L2 atomic counter + s_sleep spin.
__global__ __launch_bounds__(256) void lstm_rec_kernel(
    const float* __restrict__ pre,   // [S,B,4H]
    const bf16*  __restrict__ whh,   // [4H,H] bf16
    bf16* __restrict__ hstate,       // [2][B][H] bf16
    bf16* __restrict__ hseq,         // [B,S,H]  bf16
    unsigned* __restrict__ cnt) {
  __shared__ __align__(16) bf16  s_whh[64 * H_];        // 32 KB
  __shared__ __align__(16) float s_gates[4 * B_ * 16];  // 16 KB, [g][b*16+hu]
  __shared__ __align__(16) float s_c[B_ * 16];          //  4 KB

  const int tid = threadIdx.x;
  const int wg = blockIdx.x;            // 0..15
  const int wid = tid >> 5, lane = tid & 31;
  const int mtile = wid & 3;            // batch tile 0..3
  const int g0 = (wid >> 2) * 2;        // gate pair {0,1} or {2,3}
  const int n = lane & 15, mo = (lane >> 4) << 3;

  // Load this WG's Whh slice into LDS: rows g*H + wg*16 + r, r=0..15, g=0..3.
  {
    uint4* dst = (uint4*)s_whh;
    for (int i = tid; i < 64 * (H_ / 8); i += 256) {  // 32 uint4 per row
      int row = i >> 5, c4 = i & 31;
      int g = row >> 4, r = row & 15;
      const uint4* src = (const uint4*)(whh + (size_t)(g * H_ + wg * 16 + r) * H_);
      dst[i] = src[c4];
    }
  }
  for (int i = tid; i < B_ * 16; i += 256) s_c[i] = 0.0f;
  __syncthreads();

  for (int t = 0; t < S_; ++t) {
    // Accumulators start from the precomputed input projection + biases.
    v8f acc0, acc1;
    {
      const float* base = pre + ((size_t)t * B_ + mtile * 16) * G4H + wg * 16 + n;
      const float* c0 = base + (size_t)(g0) * H_;
      const float* c1 = base + (size_t)(g0 + 1) * H_;
      #pragma unroll
      for (int r = 0; r < 8; ++r) {
        size_t off = (size_t)(mo + r) * G4H;   // batch stride (4KB)
        acc0[r] = c0[off];
        acc1[r] = c1[off];
      }
    }
    // Prefetch next step's contiguous 256KB gate block while we compute.
    if (t + 1 < S_) {
      const char* nxt = (const char*)(pre + (size_t)(t + 1) * B_ * G4H);
      #pragma unroll
      for (int j = 0; j < 8; ++j)
        __builtin_prefetch(nxt + (size_t)tid * 1024 + j * 128, 0, 1);
    }
    if (t > 0) {
      const bf16* hbuf = hstate + (size_t)(t & 1) * B_ * H_ + (mtile * 16) * H_;
      #pragma unroll
      for (int kk = 0; kk < H_ / 32; ++kk) {
        v16bf a  = load_frag_bf16(hbuf + kk * 32, H_, lane);
        v16bf b0 = load_frag_bf16(s_whh + (g0) * 16 * H_ + kk * 32, H_, lane);
        v16bf b1 = load_frag_bf16(s_whh + (g0 + 1) * 16 * H_ + kk * 32, H_, lane);
        acc0 = wmma_bf16(a, b0, acc0);
        acc1 = wmma_bf16(a, b1, acc1);
      }
    }
    // Scatter gate pre-activations into LDS.
    #pragma unroll
    for (int r = 0; r < 8; ++r) {
      int b = mtile * 16 + mo + r;
      s_gates[(g0) * (B_ * 16) + b * 16 + n] = acc0[r];
      s_gates[(g0 + 1) * (B_ * 16) + b * 16 + n] = acc1[r];
    }
    __syncthreads();
    // LSTM cell update for this WG's 64x16 (batch, hidden) slice.
    {
      bf16* hw = hstate + (size_t)((t + 1) & 1) * B_ * H_;
      #pragma unroll
      for (int j = 0; j < 4; ++j) {
        int e = tid + j * 256;              // e = b*16 + hu_local
        int b = e >> 4, hu = e & 15;
        float ig = fast_sigmoid(s_gates[0 * (B_ * 16) + e]);
        float fg = fast_sigmoid(s_gates[1 * (B_ * 16) + e]);
        float gg = fast_tanh(s_gates[2 * (B_ * 16) + e]);
        float og = fast_sigmoid(s_gates[3 * (B_ * 16) + e]);
        float c = fg * s_c[e] + ig * gg;
        s_c[e] = c;
        float h = og * fast_tanh(c);
        int ghu = wg * 16 + hu;
        hw[b * H_ + ghu] = (bf16)h;
        hseq[((size_t)b * S_ + t) * H_ + ghu] = (bf16)h;
      }
    }
    // Grid-wide barrier (publishes the new h buffer; also guards s_gates reuse).
    __syncthreads();
    if (tid == 0) {
      __threadfence();
      atomicAdd(cnt, 1u);
      volatile unsigned* vc = cnt;
      unsigned target = (unsigned)(t + 1) * NWG;
      while (*vc < target) __builtin_amdgcn_s_sleep(2);
      __threadfence();
    }
    __syncthreads();
  }
}

// ---------------- FC head with length masking ----------------
__global__ __launch_bounds__(256) void fc_kernel(
    const bf16* __restrict__ h2, const int* __restrict__ lengths,
    const float* __restrict__ wfc, const float* __restrict__ bfc,
    float* __restrict__ out) {
  __shared__ float s_w[C_ * H_];
  __shared__ float s_b[C_];
  const int tid = threadIdx.x;
  for (int i = tid; i < C_ * H_; i += 256) s_w[i] = wfc[i];
  if (tid < C_) s_b[tid] = bfc[tid];
  __syncthreads();
  int p = blockIdx.x * 256 + tid;
  if (p >= B_ * S_) return;
  int b = p >> 11;          // / S_
  int s = p & (S_ - 1);
  float acc[C_];
  #pragma unroll
  for (int c = 0; c < C_; ++c) acc[c] = s_b[c];
  if (s < lengths[b]) {
    const bf16* hr = h2 + (size_t)p * H_;
    for (int k = 0; k < H_; k += 8) {
      union { uint4 u; bf16 h[8]; } v;
      v.u = *(const uint4*)(hr + k);
      #pragma unroll
      for (int j = 0; j < 8; ++j) {
        float hv = (float)v.h[j];
        #pragma unroll
        for (int c = 0; c < C_; ++c) acc[c] += hv * s_w[c * H_ + k + j];
      }
    }
  }
  #pragma unroll
  for (int c = 0; c < C_; ++c) out[(size_t)p * C_ + c] = acc[c];
}

// ---------------- launch ----------------
extern "C" void kernel_launch(void* const* d_in, const int* in_sizes, int n_in,
                              void* d_out, int out_size, void* d_ws, size_t ws_size,
                              hipStream_t stream) {
  const float* x    = (const float*)d_in[0];
  const int*   lens = (const int*)d_in[1];
  const float* Wih0 = (const float*)d_in[2];
  const float* Whh0 = (const float*)d_in[3];
  const float* bih0 = (const float*)d_in[4];
  const float* bhh0 = (const float*)d_in[5];
  const float* Wih1 = (const float*)d_in[6];
  const float* Whh1 = (const float*)d_in[7];
  const float* bih1 = (const float*)d_in[8];
  const float* bhh1 = (const float*)d_in[9];
  const float* Wfc  = (const float*)d_in[10];
  const float* bfc  = (const float*)d_in[11];
  float* out = (float*)d_out;

  if (ws_size < WS_NEED) return;
  char* ws = (char*)d_ws;
  float* P       = (float*)(ws + OFF_P);
  bf16*  H1      = (bf16*)(ws + OFF_H1);
  bf16*  H2      = (bf16*)(ws + OFF_H2);
  bf16*  whh0b   = (bf16*)(ws + OFF_WHH0);
  bf16*  whh1b   = (bf16*)(ws + OFF_WHH1);
  bf16*  wih0b   = (bf16*)(ws + OFF_WIH0);
  bf16*  wih1b   = (bf16*)(ws + OFF_WIH1);
  float* b0      = (float*)(ws + OFF_B0);
  float* b1      = (float*)(ws + OFF_B1);
  bf16*  hst     = (bf16*)(ws + OFF_HST);
  unsigned* cnt  = (unsigned*)(ws + OFF_CNT);

  const int nWhh = G4H * H_;   // 262144
  const int nWih0 = G4H * D_;  // 65536

  cvt_bf16_kernel<<<(nWhh + 255) / 256, 256, 0, stream>>>(Whh0, whh0b, nWhh);
  cvt_bf16_kernel<<<(nWhh + 255) / 256, 256, 0, stream>>>(Whh1, whh1b, nWhh);
  cvt_bf16_kernel<<<(nWih0 + 255) / 256, 256, 0, stream>>>(Wih0, wih0b, nWih0);
  cvt_bf16_kernel<<<(nWhh + 255) / 256, 256, 0, stream>>>(Wih1, wih1b, nWhh);
  add_bias_kernel<<<(G4H + 255) / 256, 256, 0, stream>>>(bih0, bhh0, b0, G4H);
  add_bias_kernel<<<(G4H + 255) / 256, 256, 0, stream>>>(bih1, bhh1, b1, G4H);
  zero_u32_kernel<<<1, 64, 0, stream>>>(cnt, 64);

  const int preBlocks = (B_ * S_ / 16) * 2;  // 16384 (8 waves x 4 N-tiles each)

  pre_gemm_x_kernel<<<preBlocks, 256, 0, stream>>>(x, wih0b, b0, P);
  lstm_rec_kernel<<<NWG, 256, 0, stream>>>(P, whh0b, hst, H1, cnt + 0);
  pre_gemm_h_kernel<<<preBlocks, 256, 0, stream>>>(H1, wih1b, b1, P);
  lstm_rec_kernel<<<NWG, 256, 0, stream>>>(P, whh1b, hst, H2, cnt + 16);
  fc_kernel<<<(B_ * S_ + 255) / 256, 256, 0, stream>>>(H2, lens, Wfc, bfc, out);
}
